// DiscreteSequenceModel_77232101917085
// MI455X (gfx1250) — matrix operand order repaired and checked by reading, not compiled, not run on hardware
//
#include <hip/hip_runtime.h>

// ---------------------------------------------------------------------------
// DiscreteSequenceModel (GRU scan) for MI455X / gfx1250
// B=256, T=512, H=1024, D=256
// Per step: fused GEMM  [B x (3H+3H+D)=6400 x K=H=1024]  via v_wmma_f32_16x16x32_bf16
// Weights converted once to a concatenated bf16 matrix resident in L2 (13 MB).
// Staging now uses the CDNA5 async global->LDS path (ASYNCcnt) instead of
// global_load_b128 + ds_store_b128, freeing VGPRs and the loadcnt/dscnt waits.
// ---------------------------------------------------------------------------

#define BDIM 256
#define TSTEPS 512
#define HDIM 1024
#define DDIM 256
#define KDIM 1024      // == HDIM (GEMM reduction dim)
#define NTOT 6400      // 3H + 3H + D
#define TH3 3072       // 3H

typedef __attribute__((ext_vector_type(4)))  unsigned int u32x4;
typedef __attribute__((ext_vector_type(16))) __bf16       v16bf;
typedef __attribute__((ext_vector_type(8)))  float        v8f;

union Frag { v16bf v; u32x4 q[2]; };

static __device__ __forceinline__ unsigned short f2bf(float f) {
    unsigned int u = __float_as_uint(f);
    unsigned int r = u + 0x7FFFu + ((u >> 16) & 1u);   // round-to-nearest-even
    return (unsigned short)(r >> 16);
}

// CDNA5 async copy: 16 B per lane, global -> LDS, tracked by ASYNCcnt.
// ISA 15.18.3 opcode 98: global_load_async_to_lds_b128 vdst(lds byte addr), vaddr(64b), off
static __device__ __forceinline__ void async_ld16(unsigned ldsOff, const void* gptr) {
    asm volatile("global_load_async_to_lds_b128 %0, %1, off"
                 :: "v"(ldsOff), "v"((unsigned long long)(uintptr_t)gptr)
                 : "memory");
}
static __device__ __forceinline__ void wait_async0() {
    asm volatile("s_wait_asynccnt 0x0" ::: "memory");
}

// ---------------------------------------------------------------------------
// One-time: build Wcat (bf16, rows: [w_ih | w_hh | dec_w], each K=1024 wide)
// and bias_cat (f32: [bias(3H) | zeros(3H) | dec_b(D)])
// ---------------------------------------------------------------------------
__global__ __launch_bounds__(256) void k_convert(const float* __restrict__ w_ih,
                                                 const float* __restrict__ w_hh,
                                                 const float* __restrict__ dec_w,
                                                 const float* __restrict__ bias,
                                                 const float* __restrict__ dec_b,
                                                 unsigned short* __restrict__ wcat,
                                                 float* __restrict__ bias_cat) {
    const long long idx = (long long)blockIdx.x * blockDim.x + threadIdx.x;
    const long long total = (long long)NTOT * KDIM;
    if (idx < total) {
        const int row = (int)(idx / KDIM);
        const int col = (int)(idx % KDIM);
        float s;
        if (row < TH3)            s = w_ih[(long long)row * KDIM + col];
        else if (row < 2 * TH3)   s = w_hh[(long long)(row - TH3) * KDIM + col];
        else                      s = dec_w[(long long)(row - 2 * TH3) * KDIM + col];
        wcat[idx] = f2bf(s);
    }
    if (idx < NTOT) {
        const int i = (int)idx;
        bias_cat[i] = (i < TH3) ? bias[i] : ((i < 2 * TH3) ? 0.f : dec_b[i - 2 * TH3]);
    }
}

// ---------------------------------------------------------------------------
// One-time encoder: h0 = y0 @ enc_w^T + enc_b   (67 MFLOP, negligible)
// ---------------------------------------------------------------------------
__global__ __launch_bounds__(256) void k_encoder(const float* __restrict__ y0,
                                                 const float* __restrict__ enc_w,
                                                 const float* __restrict__ enc_b,
                                                 float* __restrict__ h32,
                                                 unsigned short* __restrict__ h16) {
    const int idx = blockIdx.x * blockDim.x + threadIdx.x;   // 256*1024 threads
    const int b = idx / HDIM;
    const int j = idx % HDIM;
    const float* yr = y0 + (long long)b * DDIM;
    const float* wr = enc_w + (long long)j * DDIM;
    float s = enc_b[j];
    #pragma unroll 4
    for (int d = 0; d < DDIM; ++d) s += yr[d] * wr[d];
    h32[idx] = s;
    h16[idx] = f2bf(s);
}

// ---------------------------------------------------------------------------
// Per-step fused GEMM: C[256 x 6400] = bf16(h) @ Wcat^T (+bias_cat)
// Epilogue routes columns:  [0,3H)->gi   [3H,6H)->gh   [6H,6H+D)->preds[:,t,:]
// Block: 256 thr (8 waves), tile M=128 x N=64, K-step 32, wave does 32x32.
// ---------------------------------------------------------------------------
#define BM 128
#define BN 64
#define BK 32
#define LDS_STRIDE 20   // dwords per 32-bf16 row: 80B, 16B-aligned, conflict-free

__global__ __launch_bounds__(256) void k_step_gemm(const unsigned short* __restrict__ h16,
                                                   const unsigned short* __restrict__ wcat,
                                                   const float* __restrict__ bias_cat,
                                                   float* __restrict__ gi,
                                                   float* __restrict__ gh,
                                                   float* __restrict__ out,
                                                   int t) {
    __shared__ unsigned int ldsA[BM * LDS_STRIDE];   // 10240 B
    __shared__ unsigned int ldsB[BN * LDS_STRIDE];   //  5120 B

    const int tid   = threadIdx.x;
    const int lane  = tid & 31;
    const int wave  = tid >> 5;        // 0..7
    const int waveM = wave >> 1;       // 0..3 -> 32-row slice
    const int waveN = wave & 1;        // 0..1 -> 32-col slice
    const int kh    = lane >> 4;       // lane half (K-half selector)
    const int ml    = lane & 15;       // row/col within 16

    const int n0 = blockIdx.x * BN;    // 0..6336
    const int m0 = blockIdx.y * BM;    // 0 or 128

    v8f acc[2][2] = {};

    // per-thread staging coordinates (fixed across K loop)
    const int ldRow = tid >> 2, ldCw = tid & 3;
    const unsigned ldsAoff0 = (unsigned)(uintptr_t)&ldsA[(ldRow      ) * LDS_STRIDE + ldCw * 4];
    const unsigned ldsAoff1 = (unsigned)(uintptr_t)&ldsA[(ldRow +  64) * LDS_STRIDE + ldCw * 4];
    const unsigned ldsBoff  = (unsigned)(uintptr_t)&ldsB[(ldRow      ) * LDS_STRIDE + ldCw * 4];
    const unsigned short* gA0 = h16  + (long long)(m0 + ldRow     ) * HDIM + ldCw * 8;
    const unsigned short* gA1 = h16  + (long long)(m0 + ldRow + 64) * HDIM + ldCw * 8;
    const unsigned short* gB  = wcat + (long long)(n0 + ldRow     ) * KDIM + ldCw * 8;

    for (int kt = 0; kt < KDIM / BK; ++kt) {
        const int k0 = kt * BK;

        // Prefetch next weight K-slice (global_prefetch_b8) — weights stay hot in L2.
        if (kt + 1 < KDIM / BK)
            __builtin_prefetch(gB + k0 + BK, 0, 1);

        // Async stage A (128x32 bf16) and B (64x32 bf16) tiles: 3x 16B per thread,
        // global -> LDS directly (ASYNCcnt), no VGPR staging.
        async_ld16(ldsAoff0, gA0 + k0);
        async_ld16(ldsAoff1, gA1 + k0);
        async_ld16(ldsBoff,  gB  + k0);
        wait_async0();
        __syncthreads();

        // Gather fragments per CDNA5 16-bit WMMA VGPR layouts (2x ds_load_b128 each)
        Frag aF[2], bF[2];
        #pragma unroll
        for (int mi = 0; mi < 2; ++mi) {
            const int rb = (waveM * 32 + mi * 16 + ml) * LDS_STRIDE;
            aF[mi].q[0] = *(const u32x4*)&ldsA[rb + 4 * kh];        // K 0..7 / 8..15
            aF[mi].q[1] = *(const u32x4*)&ldsA[rb + 8 + 4 * kh];    // K 16..23 / 24..31
        }
        #pragma unroll
        for (int ni = 0; ni < 2; ++ni) {
            const int rb = (waveN * 32 + ni * 16 + ml) * LDS_STRIDE;
            bF[ni].q[0] = *(const u32x4*)&ldsB[rb + 4 * kh];        // K 0..7 / 8..15
            bF[ni].q[1] = *(const u32x4*)&ldsB[rb + 8 + 4 * kh];    // K 16..23 / 24..31
        }

        #pragma unroll
        for (int mi = 0; mi < 2; ++mi)
            #pragma unroll
            for (int ni = 0; ni < 2; ++ni)
                acc[mi][ni] = __builtin_amdgcn_wmma_f32_16x16x32_bf16(
                    false, aF[mi].v, false, bF[ni].v,
                    (short)0, acc[mi][ni], false, false);
        __syncthreads();
    }

    // Epilogue: bias add + column routing.  C layout: VGPR r -> M = r + 8*kh, N = ml.
    #pragma unroll
    for (int mi = 0; mi < 2; ++mi) {
        #pragma unroll
        for (int ni = 0; ni < 2; ++ni) {
            const int gn = n0 + waveN * 32 + ni * 16 + ml;
            const float bv = bias_cat[gn];
            #pragma unroll
            for (int r = 0; r < 8; ++r) {
                const int gb = m0 + waveM * 32 + mi * 16 + r + 8 * kh;
                const float v = acc[mi][ni][r] + bv;
                if (gn < TH3)
                    gi[(long long)gb * TH3 + gn] = v;
                else if (gn < 2 * TH3)
                    gh[(long long)gb * TH3 + (gn - TH3)] = v;
                else
                    out[((long long)gb * TSTEPS + t) * DDIM + (gn - 2 * TH3)] = v;
            }
        }
    }
}

// ---------------------------------------------------------------------------
// Per-step GRU gates:  h_new = n + z*(h - n)   (elementwise over [B,H])
// ---------------------------------------------------------------------------
__global__ __launch_bounds__(256) void k_gate(const float* __restrict__ gi,
                                              const float* __restrict__ gh,
                                              const float* __restrict__ bias_n,
                                              float* __restrict__ h32,
                                              unsigned short* __restrict__ h16) {
    const int idx = blockIdx.x * blockDim.x + threadIdx.x;   // 256*1024 threads
    const int b = idx / HDIM;
    const int j = idx % HDIM;
    const long long base = (long long)b * TH3 + j;

    const float ir  = gi[base];
    const float iz  = gi[base + HDIM];
    const float inn = gi[base + 2 * HDIM];
    const float hr  = gh[base];
    const float hz  = gh[base + HDIM];
    const float hn  = gh[base + 2 * HDIM];

    const float r = 1.f / (1.f + __expf(-(ir + hr)));
    const float z = 1.f / (1.f + __expf(-(iz + hz)));
    const float a = inn + r * (hn + bias_n[j]);
    const float e = __expf(-2.f * a);
    const float n = (1.f - e) / (1.f + e);          // tanh(a)

    const float h  = h32[idx];
    const float hv = n + z * (h - n);
    h32[idx] = hv;
    h16[idx] = f2bf(hv);
}

// ---------------------------------------------------------------------------
extern "C" void kernel_launch(void* const* d_in, const int* in_sizes, int n_in,
                              void* d_out, int out_size, void* d_ws, size_t ws_size,
                              hipStream_t stream) {
    (void)in_sizes; (void)n_in; (void)out_size; (void)ws_size;
    // inputs: 0 ts, 1 y0_batch, 2 enc_w, 3 enc_b, 4 w_ih, 5 w_hh, 6 bias,
    //         7 bias_n, 8 dec_w, 9 dec_b
    const float* y0     = (const float*)d_in[1];
    const float* enc_w  = (const float*)d_in[2];
    const float* enc_b  = (const float*)d_in[3];
    const float* w_ih   = (const float*)d_in[4];
    const float* w_hh   = (const float*)d_in[5];
    const float* bias   = (const float*)d_in[6];
    const float* bias_n = (const float*)d_in[7];
    const float* dec_w  = (const float*)d_in[8];
    const float* dec_b  = (const float*)d_in[9];
    float* out = (float*)d_out;

    // workspace carve-up (256B aligned)
    char* ws = (char*)d_ws;
    unsigned short* wcat     = (unsigned short*)(ws);                 // 13,107,200 B
    float*          bias_cat = (float*)(ws + 13107200);               //     25,600 B
    float*          h32      = (float*)(ws + 13132800);               //  1,048,576 B
    unsigned short* h16      = (unsigned short*)(ws + 14181376);      //    524,288 B
    float*          gi       = (float*)(ws + 14705664);               //  3,145,728 B
    float*          gh       = (float*)(ws + 17851392);               //  3,145,728 B
                                                                      // total ~21 MB

    // 1) one-time weight conversion + bias concat
    {
        const long long total = (long long)NTOT * KDIM;
        const int blocks = (int)((total + 255) / 256);                // 25600
        k_convert<<<blocks, 256, 0, stream>>>(w_ih, w_hh, dec_w, bias, dec_b,
                                              wcat, bias_cat);
    }
    // 2) encoder -> h0
    k_encoder<<<(BDIM * HDIM) / 256, 256, 0, stream>>>(y0, enc_w, enc_b, h32, h16);

    // 3) T sequential steps: fused WMMA GEMM + gate update
    dim3 gGemm(NTOT / BN, BDIM / BM);   // (100, 2)
    for (int t = 0; t < TSTEPS; ++t) {
        k_step_gemm<<<gGemm, 256, 0, stream>>>(h16, wcat, bias_cat, gi, gh, out, t);
        k_gate<<<(BDIM * HDIM) / 256, 256, 0, stream>>>(gi, gh, bias_n, h32, h16);
    }
}
